// CartesianTransformer_73710228734035
// MI455X (gfx1250) — matrix operand order repaired and checked by reading, not compiled, not run on hardware
//
#include <hip/hip_runtime.h>
#include <cmath>

typedef _Float16 half_t;
typedef _Float16 v16h __attribute__((ext_vector_type(16)));
typedef _Float16 v8h  __attribute__((ext_vector_type(8)));
typedef float    v8f  __attribute__((ext_vector_type(8)));

#define CB    1024      // batch
#define CN    64        // neighbors
#define CS    65        // tokens = N+1
#define CD    512
#define CH    8
#define CHD   64
#define CL    4
#define CFF   2048
#define MROWS (CB*CS)   // 66560
#define EROWS (CB*CN)   // 65536

// ---------------------------------------------------------------------------
// Generic WMMA GEMM: C[M,N] = epi(A[M,K] @ Wt[N,K]^T + bias)
// EPI: 0 = f32 out, 1 = silu -> half out, 2 = plain half out
// Block: 256 threads / 8 waves (4M x 2N); tile 128(M) x 128(N) x 32(K).
// Each wave: 32x64 output = 8 accumulators, 8 WMMAs per K-step.
// Double-buffered LDS, staged with global_load_async_to_lds_b128 (ASYNCcnt).
// ---------------------------------------------------------------------------
template <int EPI>
__global__ __launch_bounds__(256)
void gemm_wmma(const half_t* __restrict__ A, const half_t* __restrict__ Wt,
               const float* __restrict__ bias, float* __restrict__ Cf,
               half_t* __restrict__ Ch, int M, int N, int K)
{
    __shared__ half_t lA[2][128 * 32];
    __shared__ half_t lB[2][128 * 32];

    const int tid    = threadIdx.x;
    const int wave   = tid >> 5;
    const int lane   = tid & 31;
    const int wave_m = wave & 3;   // 0..3 -> 32-row strip
    const int wave_n = wave >> 2;  // 0..1 -> 64-col strip
    const int bm     = blockIdx.x * 128;
    const int bn     = blockIdx.y * 128;

    v8f acc[2][4] = {};

    // staging map: 2 threads per row, 16 halves (32B) each
    const int srow = tid >> 1;        // 0..127
    const int soff = (tid & 1) * 16;  // halves

    const int r  = lane & 15;
    const int kg = lane >> 4;

    const unsigned ldsA0 = (unsigned)(size_t)&lA[0][0];
    const unsigned ldsB0 = (unsigned)(size_t)&lB[0][0];

    const half_t* gA0 = A  + (size_t)(bm + srow) * K + soff;
    const half_t* gB0 = Wt + (size_t)(bn + srow) * K + soff;
    const unsigned lauA = (unsigned)((srow * 32 + soff) * 2);

    // async-stage one 128x32 A tile and 128x32 B tile into buffer `buf`
    auto stage_async = [&](int buf, int kt) {
        const half_t* gA = gA0 + kt;
        const half_t* gB = gB0 + kt;
        unsigned la = ldsA0 + (unsigned)(buf * 128 * 32 * 2) + lauA;
        unsigned lb = ldsB0 + (unsigned)(buf * 128 * 32 * 2) + lauA;
        asm volatile("global_load_async_to_lds_b128 %0, %1, off"
                     :: "v"(la), "v"((unsigned long long)gA) : "memory");
        asm volatile("global_load_async_to_lds_b128 %0, %1, off"
                     :: "v"(la + 16u), "v"((unsigned long long)(gA + 8)) : "memory");
        asm volatile("global_load_async_to_lds_b128 %0, %1, off"
                     :: "v"(lb), "v"((unsigned long long)gB) : "memory");
        asm volatile("global_load_async_to_lds_b128 %0, %1, off"
                     :: "v"(lb + 16u), "v"((unsigned long long)(gB + 8)) : "memory");
    };

    const int nk = K >> 5;
    stage_async(0, 0);
    asm volatile("s_wait_asynccnt 0x0" ::: "memory");
    __syncthreads();

    for (int t = 0; t < nk; ++t) {
        const int cur = t & 1;
        if (t + 1 < nk) stage_async((t + 1) & 1, (t + 1) * 32);

        const half_t* bufA = &lA[cur][0];
        const half_t* bufB = &lB[cur][0];

        // A fragments (ISA 16-bit A 16x32 layout): lane&15 = M row,
        // lane>>4 selects K half-groups {kg*8..} and {16+kg*8..}
        union { v16h v; v8h h[2]; } a0, a1;
        const int rb = wave_m * 32;
        a0.h[0] = *(const v8h*)(bufA + (rb + r) * 32 + kg * 8);
        a0.h[1] = *(const v8h*)(bufA + (rb + r) * 32 + 16 + kg * 8);
        a1.h[0] = *(const v8h*)(bufA + (rb + 16 + r) * 32 + kg * 8);
        a1.h[1] = *(const v8h*)(bufA + (rb + 16 + r) * 32 + 16 + kg * 8);

        // B fragments: lane&15 = N col, K half-groups {kg*8..} and {16+kg*8..}
        v16h bf[4];
        #pragma unroll
        for (int j = 0; j < 4; ++j) {
            union { v16h v; v8h h[2]; } bt;
            bt.h[0] = *(const v8h*)(bufB + (wave_n * 64 + j * 16 + r) * 32 + kg * 8);
            bt.h[1] = *(const v8h*)(bufB + (wave_n * 64 + j * 16 + r) * 32 + 16 + kg * 8);
            bf[j] = bt.v;
        }

        #pragma unroll
        for (int j = 0; j < 4; ++j) {
            acc[0][j] = __builtin_amdgcn_wmma_f32_16x16x32_f16(
                false, a0.v, false, bf[j], (short)0, acc[0][j], false, false);
            acc[1][j] = __builtin_amdgcn_wmma_f32_16x16x32_f16(
                false, a1.v, false, bf[j], (short)0, acc[1][j], false, false);
        }

        asm volatile("s_wait_asynccnt 0x0" ::: "memory");
        __syncthreads();
    }

    // epilogue (ISA 32-bit C/D 16x16 layout: VGPR vg -> M = vg + 8*(lane>>4))
    #pragma unroll
    for (int mi = 0; mi < 2; ++mi) {
        #pragma unroll
        for (int j = 0; j < 4; ++j) {
            const int col = bn + wave_n * 64 + j * 16 + r;
            const float bv = bias ? bias[col] : 0.0f;
            #pragma unroll
            for (int vg = 0; vg < 8; ++vg) {
                const int row = bm + wave_m * 32 + mi * 16 + kg * 8 + vg;
                float v = acc[mi][j][vg] + bv;
                if (EPI == 1) {
                    float s = v / (1.0f + __expf(-v));
                    Ch[(size_t)row * N + col] = (half_t)s;
                } else if (EPI == 2) {
                    Ch[(size_t)row * N + col] = (half_t)v;
                } else {
                    Cf[(size_t)row * N + col] = v;
                }
            }
        }
    }
}

// ---------------------------------------------------------------------------
// Weight convert + transpose: W[K][N] f32 -> Wt[N][K] half
// ---------------------------------------------------------------------------
__global__ void transpose_to_half(const float* __restrict__ W,
                                  half_t* __restrict__ Wt, int K, int N)
{
    size_t i = (size_t)blockIdx.x * 256 + threadIdx.x;
    if (i >= (size_t)K * N) return;
    int k = (int)(i / N), n = (int)(i % N);
    Wt[(size_t)n * K + k] = (half_t)W[i];
}

// ---------------------------------------------------------------------------
// Encoder: build A = concat(edge_feat, neigh_emb gather, msgs) in half
// ---------------------------------------------------------------------------
__global__ void build_encoder_A(const float* __restrict__ msgs,
                                const int* __restrict__ idxN,
                                const float* __restrict__ ev,
                                const float* __restrict__ ed,
                                const float* __restrict__ nemb,
                                const float* __restrict__ W_edge,
                                const float* __restrict__ b_edge,
                                half_t* __restrict__ Aenc)
{
    size_t i = (size_t)blockIdx.x * 256 + threadIdx.x;
    if (i >= (size_t)EROWS * (3 * CD)) return;
    size_t r = i / (3 * CD);
    int d3 = (int)(i % (3 * CD));
    float v;
    if (d3 < CD) {
        v = ev[r * 3 + 0] * W_edge[d3] + ev[r * 3 + 1] * W_edge[CD + d3] +
            ev[r * 3 + 2] * W_edge[2 * CD + d3] + ed[r] * W_edge[3 * CD + d3] +
            b_edge[d3];
    } else if (d3 < 2 * CD) {
        v = nemb[(size_t)idxN[r] * CD + (d3 - CD)];
    } else {
        v = msgs[r * CD + (d3 - 2 * CD)];
    }
    Aenc[i] = (half_t)v;
}

__global__ void build_cls(const float* __restrict__ node_emb,
                          const int* __restrict__ idx_nodes,
                          float* __restrict__ tok, half_t* __restrict__ tokh)
{
    int i = blockIdx.x * 256 + threadIdx.x;
    if (i >= CB * CD) return;
    int b = i >> 9, d = i & (CD - 1);
    float v = node_emb[(size_t)idx_nodes[b] * CD + d];
    size_t o = (size_t)b * CS * CD + d;
    tok[o] = v;
    tokh[o] = (half_t)v;
}

__global__ void assemble_tokens(const float* __restrict__ enc_out,
                                float* __restrict__ tok,
                                half_t* __restrict__ tokh)
{
    size_t i = (size_t)blockIdx.x * 256 + threadIdx.x;
    if (i >= (size_t)EROWS * CD) return;
    size_t r = i >> 9;
    int d = (int)(i & (CD - 1));
    int b = (int)(r >> 6), n = (int)(r & 63);
    size_t o = ((size_t)b * CS + 1 + n) * CD + d;
    float v = enc_out[i];
    tok[o] = v;
    tokh[o] = (half_t)v;
}

// ---------------------------------------------------------------------------
// Attention: one block per (b,h); S=65, HD=64; online softmax in VALU
// ---------------------------------------------------------------------------
__global__ __launch_bounds__(128)
void attention_kernel(const half_t* __restrict__ qkv,
                      const float* __restrict__ cutoff,
                      const unsigned char* __restrict__ pmask,
                      half_t* __restrict__ outh)
{
    const int b = blockIdx.x >> 3;
    const int h = blockIdx.x & 7;
    const int tid = threadIdx.x;
    __shared__ float sQ[CS][CHD];
    __shared__ float sK[CS][CHD];
    __shared__ float sV[CS][CHD];
    __shared__ float sMask[CS];

    for (int i = tid; i < CS * CHD; i += 128) {
        int s = i >> 6, d = i & 63;
        size_t base = ((size_t)b * CS + s) * (3 * CD) + h * CHD + d;
        sQ[s][d] = (float)qkv[base];
        sK[s][d] = (float)qkv[base + CD];
        sV[s][d] = (float)qkv[base + 2 * CD];
    }
    for (int s = tid; s < CS; s += 128) {
        float m = 0.0f;
        if (s > 0) {
            size_t j = (size_t)b * CN + (s - 1);
            float cf = pmask[j] ? cutoff[j] : 0.0f;
            m = __logf(fmaxf(cf, 1e-15f));
        }
        sMask[s] = m;
    }
    __syncthreads();

    if (tid < CS) {
        const int q = tid;
        float mx = -1e30f, l = 0.0f;
        float accd[CHD];
        #pragma unroll
        for (int d = 0; d < CHD; ++d) accd[d] = 0.0f;
        const float scale = 0.125f;  // 1/sqrt(64)
        for (int k = 0; k < CS; ++k) {
            float s = 0.0f;
            #pragma unroll
            for (int d = 0; d < CHD; ++d) s += sQ[q][d] * sK[k][d];
            s = s * scale + sMask[k];
            float nm = fmaxf(mx, s);
            float corr = __expf(mx - nm);
            float p = __expf(s - nm);
            l = l * corr + p;
            #pragma unroll
            for (int d = 0; d < CHD; ++d) accd[d] = accd[d] * corr + p * sV[k][d];
            mx = nm;
        }
        float inv = 1.0f / l;
        size_t base = ((size_t)b * CS + q) * CD + h * CHD;
        for (int d = 0; d < CHD; ++d) outh[base + d] = (half_t)(accd[d] * inv);
    }
}

// ---------------------------------------------------------------------------
// Residual add + LayerNorm; one wave32 per row of 512
// ---------------------------------------------------------------------------
__global__ __launch_bounds__(256)
void add_ln_kernel(float* __restrict__ tok, half_t* __restrict__ tokh,
                   const float* __restrict__ delta,
                   const float* __restrict__ g, const float* __restrict__ be,
                   int nrows)
{
    const int wave = threadIdx.x >> 5;
    const int lane = threadIdx.x & 31;
    const int row = blockIdx.x * 8 + wave;
    if (row >= nrows) return;
    float* t = tok + (size_t)row * CD;
    const float* dlt = delta + (size_t)row * CD;
    float x[16];
    float sum = 0.f;
    #pragma unroll
    for (int i = 0; i < 16; ++i) {
        x[i] = t[lane + i * 32] + dlt[lane + i * 32];
        sum += x[i];
    }
    #pragma unroll
    for (int off = 16; off >= 1; off >>= 1) sum += __shfl_xor(sum, off, 32);
    float mean = sum * (1.0f / CD);
    float var = 0.f;
    #pragma unroll
    for (int i = 0; i < 16; ++i) { float d = x[i] - mean; var += d * d; }
    #pragma unroll
    for (int off = 16; off >= 1; off >>= 1) var += __shfl_xor(var, off, 32);
    float rstd = rsqrtf(var * (1.0f / CD) + 1e-5f);
    half_t* th = tokh + (size_t)row * CD;
    #pragma unroll
    for (int i = 0; i < 16; ++i) {
        int c = lane + i * 32;
        float y = (x[i] - mean) * rstd * g[c] + be[c];
        t[c] = y;
        th[c] = (half_t)y;
    }
}

__global__ void write_output(const float* __restrict__ tok, float* __restrict__ out)
{
    size_t i = (size_t)blockIdx.x * 256 + threadIdx.x;
    const size_t clsN = (size_t)CB * CD;
    const size_t total = clsN + (size_t)EROWS * CD;
    if (i >= total) return;
    if (i < clsN) {
        int b = (int)(i >> 9), d = (int)(i & (CD - 1));
        out[i] = tok[((size_t)b * CS) * CD + d];
    } else {
        size_t j = i - clsN;
        size_t r = j >> 9;
        int d = (int)(j & (CD - 1));
        int b = (int)(r >> 6), n = (int)(r & 63);
        out[i] = tok[((size_t)b * CS + 1 + n) * CD + d];
    }
}

// ---------------------------------------------------------------------------
static inline int ceil_div_i(size_t a, int b) { return (int)((a + b - 1) / b); }

extern "C" void kernel_launch(void* const* d_in, const int* in_sizes, int n_in,
                              void* d_out, int out_size, void* d_ws, size_t ws_size,
                              hipStream_t stream)
{
    (void)in_sizes; (void)n_in; (void)out_size; (void)ws_size;
    const float* msgs    = (const float*)d_in[0];
    const int*   idxNode = (const int*)d_in[1];
    const int*   idxNgh  = (const int*)d_in[2];
    const float* ev      = (const float*)d_in[3];
    const unsigned char* pmask = (const unsigned char*)d_in[4];
    const float* ed      = (const float*)d_in[5];
    const float* cutoff  = (const float*)d_in[6];
    const float* node_emb = (const float*)d_in[7];
    const float* neigh_emb = (const float*)d_in[8];
    const float* W_edge = (const float*)d_in[9];
    const float* b_edge = (const float*)d_in[10];
    const float* W_c1 = (const float*)d_in[11];
    const float* b_c1 = (const float*)d_in[12];
    const float* W_c2 = (const float*)d_in[13];
    const float* b_c2 = (const float*)d_in[14];
    const float* Wqkv = (const float*)d_in[15];
    const float* bqkv = (const float*)d_in[16];
    const float* Wo   = (const float*)d_in[17];
    const float* bo   = (const float*)d_in[18];
    const float* g_attn  = (const float*)d_in[19];
    const float* be_attn = (const float*)d_in[20];
    const float* W_m1 = (const float*)d_in[21];
    const float* b_m1 = (const float*)d_in[22];
    const float* W_m2 = (const float*)d_in[23];
    const float* b_m2 = (const float*)d_in[24];
    const float* g_mlp  = (const float*)d_in[25];
    const float* be_mlp = (const float*)d_in[26];
    float* out = (float*)d_out;

    // ---- workspace carve-up ----
    size_t off = 0;
    char* base = (char*)d_ws;
    auto carve = [&](size_t bytes) -> char* {
        char* p = base + off;
        off += (bytes + 255) & ~(size_t)255;
        return p;
    };
    float*  tok_f = (float*)carve((size_t)MROWS * CD * 4);
    half_t* tok_h = (half_t*)carve((size_t)MROWS * CD * 2);
    half_t* big   = (half_t*)carve((size_t)MROWS * CFF * 2);   // encA / qkv / mlp hidden
    float*  gbuf  = (float*)carve((size_t)MROWS * CD * 4);     // GEMM f32 out
    half_t* atth  = (half_t*)carve((size_t)MROWS * CD * 2);    // attn out / enc hidden
    half_t* Wc1t  = (half_t*)carve((size_t)3 * CD * CD * 2);
    half_t* Wc2t  = (half_t*)carve((size_t)CD * CD * 2);
    half_t* Wqkvt = (half_t*)carve((size_t)CL * CD * 3 * CD * 2);
    half_t* Wot   = (half_t*)carve((size_t)CL * CD * CD * 2);
    half_t* Wm1t  = (half_t*)carve((size_t)CL * CD * CFF * 2);
    half_t* Wm2t  = (half_t*)carve((size_t)CL * CFF * CD * 2);

    // ---- weight convert + transpose ----
    auto tpose = [&](const float* W, half_t* Wt, int K, int N) {
        size_t e = (size_t)K * N;
        hipLaunchKernelGGL(transpose_to_half, dim3(ceil_div_i(e, 256)), dim3(256),
                           0, stream, W, Wt, K, N);
    };
    tpose(W_c1, Wc1t, 3 * CD, CD);
    tpose(W_c2, Wc2t, CD, CD);
    for (int l = 0; l < CL; ++l) {
        tpose(Wqkv + (size_t)l * CD * 3 * CD, Wqkvt + (size_t)l * 3 * CD * CD, CD, 3 * CD);
        tpose(Wo   + (size_t)l * CD * CD,     Wot   + (size_t)l * CD * CD,     CD, CD);
        tpose(W_m1 + (size_t)l * CD * CFF,    Wm1t  + (size_t)l * CFF * CD,    CD, CFF);
        tpose(W_m2 + (size_t)l * CFF * CD,    Wm2t  + (size_t)l * CD * CFF,    CFF, CD);
    }

    // ---- encoder ----
    {
        size_t e = (size_t)EROWS * 3 * CD;
        hipLaunchKernelGGL(build_encoder_A, dim3(ceil_div_i(e, 256)), dim3(256), 0,
                           stream, msgs, idxNgh, ev, ed, neigh_emb, W_edge, b_edge, big);
    }
    hipLaunchKernelGGL(build_cls, dim3(ceil_div_i((size_t)CB * CD, 256)), dim3(256),
                       0, stream, node_emb, idxNode, tok_f, tok_h);
    // hidden = silu(A @ W_c1 + b_c1)
    hipLaunchKernelGGL((gemm_wmma<1>), dim3(EROWS / 128, CD / 128), dim3(256), 0, stream,
                       big, Wc1t, b_c1, (float*)nullptr, atth, EROWS, CD, 3 * CD);
    // enc_out = hidden @ W_c2 + b_c2
    hipLaunchKernelGGL((gemm_wmma<0>), dim3(EROWS / 128, CD / 128), dim3(256), 0, stream,
                       atth, Wc2t, b_c2, gbuf, (half_t*)nullptr, EROWS, CD, CD);
    hipLaunchKernelGGL(assemble_tokens, dim3(ceil_div_i((size_t)EROWS * CD, 256)),
                       dim3(256), 0, stream, gbuf, tok_f, tok_h);

    // ---- transformer layers ----
    for (int l = 0; l < CL; ++l) {
        // qkv (half out, consumed by attention)
        hipLaunchKernelGGL((gemm_wmma<2>), dim3(MROWS / 128, (3 * CD) / 128), dim3(256),
                           0, stream, tok_h, Wqkvt + (size_t)l * 3 * CD * CD,
                           bqkv + (size_t)l * 3 * CD, (float*)nullptr, big,
                           MROWS, 3 * CD, CD);
        hipLaunchKernelGGL(attention_kernel, dim3(CB * CH), dim3(128), 0, stream,
                           big, cutoff, pmask, atth);
        hipLaunchKernelGGL((gemm_wmma<0>), dim3(MROWS / 128, CD / 128), dim3(256), 0,
                           stream, atth, Wot + (size_t)l * CD * CD,
                           bo + (size_t)l * CD, gbuf, (half_t*)nullptr, MROWS, CD, CD);
        hipLaunchKernelGGL(add_ln_kernel, dim3(MROWS / 8), dim3(256), 0, stream,
                           tok_f, tok_h, gbuf, g_attn + (size_t)l * CD,
                           be_attn + (size_t)l * CD, MROWS);
        // MLP
        hipLaunchKernelGGL((gemm_wmma<1>), dim3(MROWS / 128, CFF / 128), dim3(256), 0,
                           stream, tok_h, Wm1t + (size_t)l * CFF * CD,
                           b_m1 + (size_t)l * CFF, (float*)nullptr, big, MROWS, CFF, CD);
        hipLaunchKernelGGL((gemm_wmma<0>), dim3(MROWS / 128, CD / 128), dim3(256), 0,
                           stream, big, Wm2t + (size_t)l * CD * CFF,
                           b_m2 + (size_t)l * CD, gbuf, (half_t*)nullptr, MROWS, CD, CFF);
        hipLaunchKernelGGL(add_ln_kernel, dim3(MROWS / 8), dim3(256), 0, stream,
                           tok_f, tok_h, gbuf, g_mlp + (size_t)l * CD,
                           be_mlp + (size_t)l * CD, MROWS);
    }

    // ---- outputs ----
    {
        size_t total = (size_t)CB * CD + (size_t)EROWS * CD;
        hipLaunchKernelGGL(write_output, dim3(ceil_div_i(total, 256)), dim3(256), 0,
                           stream, tok_f, out);
    }
}
